// cross_scale_trans_42219528519990
// MI455X (gfx1250) — matrix operand order repaired and checked by reading, not compiled, not run on hardware
//
#include <hip/hip_runtime.h>
#include <hip/hip_bf16.h>
#include <math.h>

typedef __attribute__((ext_vector_type(16))) _Float16 v16h;
typedef __attribute__((ext_vector_type(8)))  float    v8f;

#define EPSF 1e-5f
// 1/sqrt(1 + 1e-5): eval-mode BatchNorm with running stats (0,1)
#define BN_RSQRT 0.9999950000374997f

__device__ __forceinline__ v16h pack16(float4 a, float4 b, float4 c, float4 d)
{
  v16h r;
  r[0] = (_Float16)a.x;  r[1] = (_Float16)a.y;  r[2] = (_Float16)a.z;  r[3] = (_Float16)a.w;
  r[4] = (_Float16)b.x;  r[5] = (_Float16)b.y;  r[6] = (_Float16)b.z;  r[7] = (_Float16)b.w;
  r[8] = (_Float16)c.x;  r[9] = (_Float16)c.y;  r[10] = (_Float16)c.z; r[11] = (_Float16)c.w;
  r[12] = (_Float16)d.x; r[13] = (_Float16)d.y; r[14] = (_Float16)d.z; r[15] = (_Float16)d.w;
  return r;
}

// ---------------------------------------------------------------------------
// WMMA GEMM: Y[N,M] = A[N,K] @ W[K,M] + bias. K, M, EPI compile-time so all
// tile addresses fold to base + immediate offsets. One wave32 per 32x16
// output (two 16x16 row tiles sharing the B fragment), f32->f16 convert,
// f32 accumulate via v_wmma_f32_16x16x32_f16. Requires N % 16 == 0.
// EPI: 0 = bias, 1 = bias+ReLU, 2 = bias + BN(scale/shift) + ReLU
// ---------------------------------------------------------------------------
template<int K, int M, int EPI>
__global__ __launch_bounds__(32) void gemm_wmma_kernel(
    const float* __restrict__ A, const float* __restrict__ W,
    const float* __restrict__ bias, float* __restrict__ Y, int N,
    const float* __restrict__ bn_g, const float* __restrict__ bn_b)
{
  const int lane = threadIdx.x;
  const int hi   = lane >> 4;        // lane half (0: lanes 0-15, 1: lanes 16-31)
  const int l15  = lane & 15;
  const int row0 = blockIdx.x * 32;  // two 16-row tiles
  const int bcol = blockIdx.y * 16 + l15;
  const bool t1v = (row0 + 16) < N;  // second tile valid (N % 16 == 0)
  const float* __restrict__ A0 = A + (size_t)(row0 + l15) * K;
  const float* __restrict__ A1 = A0 + (size_t)16 * K;
  const float4 z4 = make_float4(0.f, 0.f, 0.f, 0.f);

  v8f acc0 = {}, acc1 = {};

  if constexpr (K >= 32) {
    for (int ks = 0; ks < K / 32; ++ks) {
      const int k0 = ks * 32;
      // B fragment (32x16): col = lane&15; lanes 0-15 hold K {0..7,16..23},
      // lanes 16-31 hold K {8..15,24..31}. Immediate-offset load clause.
      const float* __restrict__ Wp = W + (size_t)(k0 + hi * 8) * M + bcol;
      float bt[16];
#pragma unroll
      for (int h = 0; h < 16; ++h) bt[h] = Wp[(h < 8 ? h : h + 8) * M];
      v16h b;
#pragma unroll
      for (int h = 0; h < 16; ++h) b[h] = (_Float16)bt[h];

      // A fragments (16x32): row = lane&15; lanes 0-15 hold K {0..7,16..23},
      // lanes 16-31 hold K {8..15,24..31} -> two contiguous float4 pairs.
      const float4 a00 = *(const float4*)(A0 + k0 + hi * 8);
      const float4 a01 = *(const float4*)(A0 + k0 + hi * 8 + 4);
      const float4 a02 = *(const float4*)(A0 + k0 + 16 + hi * 8);
      const float4 a03 = *(const float4*)(A0 + k0 + 16 + hi * 8 + 4);
      float4 a10 = z4, a11 = z4, a12 = z4, a13 = z4;
      if (t1v) {
        a10 = *(const float4*)(A1 + k0 + hi * 8);
        a11 = *(const float4*)(A1 + k0 + hi * 8 + 4);
        a12 = *(const float4*)(A1 + k0 + 16 + hi * 8);
        a13 = *(const float4*)(A1 + k0 + 16 + hi * 8 + 4);
      }
      const v16h a0 = pack16(a00, a01, a02, a03);
      const v16h a1 = pack16(a10, a11, a12, a13);

      if (ks + 1 < K / 32) __builtin_prefetch(Wp + 32 * M, 0, 1);

      acc0 = __builtin_amdgcn_wmma_f32_16x16x32_f16(false, a0, false, b,
                                                    (short)0, acc0, false, false);
      acc1 = __builtin_amdgcn_wmma_f32_16x16x32_f16(false, a1, false, b,
                                                    (short)0, acc1, false, false);
    }
  }
  if constexpr ((K % 32) == 16) {      // only K == 16 here
    constexpr int k0 = (K / 32) * 32;
    float bt[16];
    const float* __restrict__ Wp = W + (size_t)(k0 + hi * 8) * M + bcol;
#pragma unroll
    for (int h = 0; h < 16; ++h) bt[h] = (h < 8) ? Wp[h * M] : 0.0f;
    v16h b;
#pragma unroll
    for (int h = 0; h < 16; ++h) b[h] = (_Float16)bt[h];

    const float4 a00 = *(const float4*)(A0 + k0 + hi * 8);
    const float4 a01 = *(const float4*)(A0 + k0 + hi * 8 + 4);
    float4 a10 = z4, a11 = z4;
    if (t1v) {
      a10 = *(const float4*)(A1 + k0 + hi * 8);
      a11 = *(const float4*)(A1 + k0 + hi * 8 + 4);
    }
    const v16h a0 = pack16(a00, a01, z4, z4);
    const v16h a1 = pack16(a10, a11, z4, z4);
    acc0 = __builtin_amdgcn_wmma_f32_16x16x32_f16(false, a0, false, b,
                                                  (short)0, acc0, false, false);
    acc1 = __builtin_amdgcn_wmma_f32_16x16x32_f16(false, a1, false, b,
                                                  (short)0, acc1, false, false);
  }

  const float bv = bias[bcol];
  float gv = 0.f, b2v = 0.f;
  if constexpr (EPI == 2) { gv = bn_g[bcol]; b2v = bn_b[bcol]; }

  // D layout: VGPR j -> row j (lanes 0-15) / j+8 (lanes 16-31), col = lane&15
  float* __restrict__ Yp = Y + (size_t)(row0 + hi * 8) * M + bcol;
#pragma unroll
  for (int j = 0; j < 8; ++j) {
    float v = acc0[j] + bv;
    if constexpr (EPI == 1)      v = fmaxf(v, 0.0f);
    else if constexpr (EPI == 2) v = fmaxf(v * BN_RSQRT * gv + b2v, 0.0f);
    Yp[(size_t)j * M] = v;
  }
  if (t1v) {
    float* __restrict__ Yq = Yp + (size_t)16 * M;
#pragma unroll
    for (int j = 0; j < 8; ++j) {
      float v = acc1[j] + bv;
      if constexpr (EPI == 1)      v = fmaxf(v, 0.0f);
      else if constexpr (EPI == 2) v = fmaxf(v * BN_RSQRT * gv + b2v, 0.0f);
      Yq[(size_t)j * M] = v;
    }
  }
}

// ---------------------------------------------------------------------------
// GroupNorm(groups=C over 256 channels) + positional-encoding MLP add, in place
// on src[N,256]. One 256-thread block per row.
// ---------------------------------------------------------------------------
__global__ __launch_bounds__(256) void gn_pe_kernel(
    float* __restrict__ src, const int* __restrict__ idx,
    const float* __restrict__ gn_g, const float* __restrict__ gn_b,
    const float* __restrict__ w1, const float* __restrict__ b1,
    const float* __restrict__ w2, const float* __restrict__ b2,
    int N, int C, int D, int H, int Wd)
{
  __shared__ float hbuf[128];
  __shared__ float xbuf[256];
  const int n = blockIdx.x;
  const int c = threadIdx.x;

  const float vz = (float)idx[n * 4 + 1] / (float)(D - 1);
  const float vy = (float)idx[n * 4 + 2] / (float)(H - 1);
  const float vx = (float)idx[n * 4 + 3] / (float)(Wd - 1);

  if (c < 128) {
    float s = b1[c] + vz * w1[0 * 128 + c] + vy * w1[1 * 128 + c] + vx * w1[2 * 128 + c];
    hbuf[c] = fmaxf(s, 0.0f);
  }
  xbuf[c] = src[(size_t)n * 256 + c];
  __syncthreads();

  float pe = b2[c];
  for (int j = 0; j < 128; ++j) pe += hbuf[j] * w2[j * 256 + c];

  const int GS = 256 / C;            // group size: 16, 8, 4, 4
  const int base = (c / GS) * GS;
  float mu = 0.0f;
  for (int j = 0; j < GS; ++j) mu += xbuf[base + j];
  mu /= (float)GS;
  float var = 0.0f;
  for (int j = 0; j < GS; ++j) { float d = xbuf[base + j] - mu; var += d * d; }
  var /= (float)GS;
  float xn = (xbuf[c] - mu) * rsqrtf(var + EPSF);
  src[(size_t)n * 256 + c] = xn * gn_g[c] + gn_b[c] + pe;
}

// ---------------------------------------------------------------------------
// Offsets: sigmoid(so)/2 in place; attn weights: softmax over P=8 per head,
// in place. One 128-thread block per row (96 offset lanes + 32 weight lanes).
// ---------------------------------------------------------------------------
__global__ __launch_bounds__(128) void act_off_attw_kernel(
    float* __restrict__ so, float* __restrict__ aw, int N)
{
  __shared__ float awr[32];
  const int n = blockIdx.x;
  const int t = threadIdx.x;
  float sv = 0.0f;
  if (t < 96) sv = so[(size_t)n * 96 + t];
  else        awr[t - 96] = aw[(size_t)n * 32 + (t - 96)];
  __syncthreads();
  if (t < 96) {
    so[(size_t)n * 96 + t] = 0.5f / (1.0f + __expf(-sv));
  } else {
    const int j = t - 96;
    const int h = j >> 3;
    float m = -1e30f;
    for (int p = 0; p < 8; ++p) m = fmaxf(m, awr[h * 8 + p]);
    float s = 0.0f;
    for (int p = 0; p < 8; ++p) s += __expf(awr[h * 8 + p] - m);
    aw[(size_t)n * 32 + j] = __expf(awr[j] - m) / s;
  }
}

// ---------------------------------------------------------------------------
__global__ void fill4_kernel(float4* __restrict__ p, size_t n4, float v)
{
  const float4 q = make_float4(v, v, v, v);
  size_t i = (size_t)blockIdx.x * blockDim.x + threadIdx.x;
  const size_t stride = (size_t)gridDim.x * blockDim.x;
  for (; i < n4; i += stride) p[i] = q;
}

// Scatter value[N,C] into voxel-major dense[D*H*W, C] (float4 packets).
__global__ void scatter_kernel(float4* __restrict__ dense,
                               const float4* __restrict__ value,
                               const int* __restrict__ idx,
                               int N, int C4, int H, int Wd)
{
  const int i = blockIdx.x * blockDim.x + threadIdx.x;
  if (i >= N * C4) return;
  const int n = i / C4, c = i % C4;
  const int z = idx[n * 4 + 1], y = idx[n * 4 + 2], x = idx[n * 4 + 3];
  const size_t lin = ((size_t)z * H + y) * Wd + x;
  dense[lin * C4 + c] = value[(size_t)n * C4 + c];
}

// ---------------------------------------------------------------------------
// Deformable trilinear sampling + attention-weighted reduction. One wave32 per
// row: all 32 lanes precompute the 32 sample corners (heads*P = 4*8) into LDS,
// lanes 0..C/4-1 then gather float4 channel packets (channel-contiguous dense
// layout -> coalesced 16B segments) and accumulate all samples.
// ---------------------------------------------------------------------------
__global__ __launch_bounds__(32) void sample_attn_kernel(
    float* __restrict__ out, const float* __restrict__ dense,
    const float* __restrict__ off, const float* __restrict__ attw,
    const int* __restrict__ idx, int N, int C, int D, int H, int Wd)
{
  __shared__ int   z0s[32], y0s[32], x0s[32];
  __shared__ float fzs[32], fys[32], fxs[32], wts[32];
  const int n = blockIdx.x;
  const int t = threadIdx.x;

  {
    const float vz = (float)idx[n * 4 + 1] / (float)(D - 1);
    const float vy = (float)idx[n * 4 + 2] / (float)(H - 1);
    const float vx = (float)idx[n * 4 + 3] / (float)(Wd - 1);
    const float* o = off + (size_t)n * 96 + t * 3;
    // grid in [-1,1] (d,h,w); align_corners=True mapping
    const float gz = 2.0f * (o[0] + vz) - 1.0f;
    const float gy = 2.0f * (o[1] + vy) - 1.0f;
    const float gx = 2.0f * (o[2] + vx) - 1.0f;
    const float z = (gz + 1.0f) * 0.5f * (float)(D - 1);
    const float y = (gy + 1.0f) * 0.5f * (float)(H - 1);
    const float x = (gx + 1.0f) * 0.5f * (float)(Wd - 1);
    const float zf = floorf(z), yf = floorf(y), xf = floorf(x);
    z0s[t] = (int)zf; y0s[t] = (int)yf; x0s[t] = (int)xf;
    fzs[t] = z - zf;  fys[t] = y - yf;  fxs[t] = x - xf;
    wts[t] = attw[(size_t)n * 32 + t];
  }
  __syncthreads();
  const int C4 = C >> 2;
  if (t >= C4) return;

  const float4* __restrict__ dense4 = (const float4*)dense;
  float ax = 0.f, ay = 0.f, az = 0.f, aww = 0.f;
  for (int s = 0; s < 32; ++s) {
    const int z0 = z0s[s], y0 = y0s[s], x0 = x0s[s];
    const float fz = fzs[s], fy = fys[s], fx = fxs[s], wa = wts[s];
#pragma unroll
    for (int dz = 0; dz < 2; ++dz)
#pragma unroll
      for (int dy = 0; dy < 2; ++dy)
#pragma unroll
        for (int dx = 0; dx < 2; ++dx) {
          const int zi = z0 + dz, yi = y0 + dy, xi = x0 + dx;
          if (zi < 0 || zi >= D || yi < 0 || yi >= H || xi < 0 || xi >= Wd) continue;
          const float w = (dz ? fz : 1.0f - fz) * (dy ? fy : 1.0f - fy) *
                          (dx ? fx : 1.0f - fx) * wa;
          const float4 v = dense4[(((size_t)zi * H + yi) * Wd + xi) * C4 + t];
          ax += v.x * w; ay += v.y * w; az += v.z * w; aww += v.w * w;
        }
  }
  ((float4*)out)[(size_t)n * C4 + t] = make_float4(ax, ay, az, aww);
}

// ---------------------------------------------------------------------------
// Y = LayerNorm(A + B) over C channels. One 64-thread block per row.
// ---------------------------------------------------------------------------
__global__ __launch_bounds__(64) void residual_ln_kernel(
    const float* __restrict__ A, const float* __restrict__ B,
    const float* __restrict__ g, const float* __restrict__ b,
    float* __restrict__ Y, int N, int C)
{
  __shared__ float xb[64];
  const int n = blockIdx.x, c = threadIdx.x;
  float x = 0.0f;
  if (c < C) { x = A[(size_t)n * C + c] + B[(size_t)n * C + c]; xb[c] = x; }
  __syncthreads();
  if (c >= C) return;
  float mu = 0.0f;
  for (int j = 0; j < C; ++j) mu += xb[j];
  mu /= (float)C;
  float var = 0.0f;
  for (int j = 0; j < C; ++j) { float d = xb[j] - mu; var += d * d; }
  var /= (float)C;
  Y[(size_t)n * C + c] = (x - mu) * rsqrtf(var + EPSF) * g[c] + b[c];
}

// cat[N,2C] = [features | tgt], float4 packets
__global__ void concat4_kernel(const float4* __restrict__ F, const float4* __restrict__ T,
                               float4* __restrict__ cat, int N, int C4)
{
  const int i = blockIdx.x * blockDim.x + threadIdx.x;
  if (i >= N * 2 * C4) return;
  const int n = i / (2 * C4), j = i % (2 * C4);
  cat[i] = (j < C4) ? F[(size_t)n * C4 + j] : T[(size_t)n * C4 + (j - C4)];
}

// ---------------------------------------------------------------------------
struct LvlP {
  const float *ipW, *ipb, *gng, *gnb, *soW, *soB, *awW, *awB, *qpW, *qpb,
              *vpW, *vpb, *opW, *opb, *lng, *lnb, *l1W, *l1b, *l2W, *l2b,
              *l3g, *l3b, *fuW, *fub, *bng, *bnb;
};
struct Bufs { float *R, *query, *value, *attn, *t1, *t2, *t3, *sob, *awb, *catb; };

template<int K, int M, int EPI>
static inline void launch_gemm(const float* A, const float* W, const float* bias,
                               float* Y, int N, const float* g, const float* b,
                               hipStream_t st)
{
  dim3 grid((unsigned)((N + 31) / 32), (unsigned)(M / 16));
  gemm_wmma_kernel<K, M, EPI><<<grid, 32, 0, st>>>(A, W, bias, Y, N, g, b);
}

template<int C>
static void run_level(const float* feats, const int* idx, const LvlP& p,
                      const float* pe1W, const float* pe1b,
                      const float* pe2W, const float* pe2b,
                      float* out, int N, int D, int H, int Wd,
                      const Bufs& B, hipStream_t st)
{
  const size_t V = (size_t)D * H * Wd;
  // 1) input projection -> src[N,256], then GroupNorm + PE MLP (in place)
  float* src = B.R;
  launch_gemm<C, 256, 0>(feats, p.ipW, p.ipb, src, N, nullptr, nullptr, st);
  gn_pe_kernel<<<N, 256, 0, st>>>(src, idx, p.gng, p.gnb, pe1W, pe1b, pe2W, pe2b,
                                  N, C, D, H, Wd);
  // 2) query / value projections
  launch_gemm<256, C, 0>(src, p.qpW, p.qpb, B.query, N, nullptr, nullptr, st);
  launch_gemm<256, C, 0>(src, p.vpW, p.vpb, B.value, N, nullptr, nullptr, st);
  // 3) sampling offsets + attention weights
  launch_gemm<C, 96, 0>(B.query, p.soW, p.soB, B.sob, N, nullptr, nullptr, st);
  launch_gemm<C, 32, 0>(B.query, p.awW, p.awB, B.awb, N, nullptr, nullptr, st);
  act_off_attw_kernel<<<N, 128, 0, st>>>(B.sob, B.awb, N);
  // 4) scatter into dense voxel-major volume (reuses R; src is dead)
  float* dense = B.R;
  fill4_kernel<<<4096, 256, 0, st>>>((float4*)dense, V * (size_t)C / 4, 1e-6f);
  scatter_kernel<<<(N * (C / 4) + 255) / 256, 256, 0, st>>>(
      (float4*)dense, (const float4*)B.value, idx, N, C / 4, H, Wd);
  // 5) deformable trilinear sampling + weighted sum -> attn[N,C]
  sample_attn_kernel<<<N, 32, 0, st>>>(B.attn, dense, B.sob, B.awb, idx, N, C, D, H, Wd);
  // 6) output projection + residual LayerNorm
  launch_gemm<C, C, 0>(B.attn, p.opW, p.opb, B.t1, N, nullptr, nullptr, st);
  residual_ln_kernel<<<N, 64, 0, st>>>(B.query, B.t1, p.lng, p.lnb, B.t2, N, C);
  // 7) FFN (hidden reuses R; dense is dead) + residual LayerNorm
  launch_gemm<C, 512, 1>(B.t2, p.l1W, p.l1b, B.R, N, nullptr, nullptr, st);
  launch_gemm<512, C, 0>(B.R, p.l2W, p.l2b, B.t1, N, nullptr, nullptr, st);
  residual_ln_kernel<<<N, 64, 0, st>>>(B.t2, B.t1, p.l3g, p.l3b, B.t3, N, C);
  // 8) fusion: Linear(concat) + eval BatchNorm + ReLU -> d_out slice
  concat4_kernel<<<(N * (2 * C / 4) + 255) / 256, 256, 0, st>>>(
      (const float4*)feats, (const float4*)B.t3, (float4*)B.catb, N, C / 4);
  launch_gemm<2 * C, C, 2>(B.catb, p.fuW, p.fub, out, N, p.bng, p.bnb, st);
}

// ---------------------------------------------------------------------------
extern "C" void kernel_launch(void* const* d_in, const int* in_sizes, int n_in,
                              void* d_out, int out_size, void* d_ws, size_t ws_size,
                              hipStream_t stream)
{
  (void)in_sizes; (void)n_in; (void)out_size; (void)ws_size;

  static const int NVOXC[4] = {60000, 40000, 20000, 10000};
  static const int SHP[4][3] = {{41, 320, 288}, {21, 160, 144}, {11, 80, 72}, {6, 40, 36}};

  float* ws = (float*)d_ws;
  // Workspace (floats). Region R is time-shared: src (15.36M) -> dense (60.46M
  // at L0) -> FFN hidden (30.72M); live ranges do not overlap.
  const size_t Rsz   = 60456960;  // max over levels of max(N*256, D*H*W*C, N*512)
  const size_t qsz   = 1280000;   // max N*C
  const size_t sosz  = 5760000;   // max N*96
  const size_t awsz  = 1920000;   // max N*32
  Bufs B;
  B.R     = ws;
  B.query = B.R + Rsz;
  B.value = B.query + qsz;
  B.attn  = B.value + qsz;
  B.t1    = B.attn + qsz;
  B.t2    = B.t1 + qsz;
  B.t3    = B.t2 + qsz;
  B.sob   = B.t3 + qsz;
  B.awb   = B.sob + sosz;
  B.catb  = B.awb + awsz;

  // params flattened in setup_inputs() insertion order:
  // d_in[0..7]  = features_l0, indices_l0, ..., features_l3, indices_l3
  // d_in[8..11] = pe.l1.W, pe.l1.b, pe.l2.W, pe.l2.b
  // d_in[12 + 26*lvl + j] = level leaves (ip, gn, so, aw, qp, vp, op, ln, l1,
  //                          l2, ln3, fu, bn; W before b)
  const float* pe1W = (const float*)d_in[8];
  const float* pe1b = (const float*)d_in[9];
  const float* pe2W = (const float*)d_in[10];
  const float* pe2b = (const float*)d_in[11];

  size_t outOff = 0;
  for (int lvl = 0; lvl < 4; ++lvl) {
    const int N = NVOXC[lvl];
    const int D = SHP[lvl][0], H = SHP[lvl][1], Wd = SHP[lvl][2];
    const float* feats = (const float*)d_in[2 * lvl];
    const int*   idx   = (const int*)d_in[2 * lvl + 1];
    const int pb = 12 + 26 * lvl;
    auto P = [&](int j) { return (const float*)d_in[pb + j]; };
    LvlP p;
    p.ipW = P(0);  p.ipb = P(1);  p.gng = P(2);  p.gnb = P(3);
    p.soW = P(4);  p.soB = P(5);  p.awW = P(6);  p.awB = P(7);
    p.qpW = P(8);  p.qpb = P(9);  p.vpW = P(10); p.vpb = P(11);
    p.opW = P(12); p.opb = P(13); p.lng = P(14); p.lnb = P(15);
    p.l1W = P(16); p.l1b = P(17); p.l2W = P(18); p.l2b = P(19);
    p.l3g = P(20); p.l3b = P(21); p.fuW = P(22); p.fub = P(23);
    p.bng = P(24); p.bnb = P(25);

    float* out = (float*)d_out + outOff;
    if (lvl == 0) {
      run_level<16>(feats, idx, p, pe1W, pe1b, pe2W, pe2b, out, N, D, H, Wd, B, stream);
      outOff += (size_t)N * 16;
    } else if (lvl == 1) {
      run_level<32>(feats, idx, p, pe1W, pe1b, pe2W, pe2b, out, N, D, H, Wd, B, stream);
      outOff += (size_t)N * 32;
    } else {
      run_level<64>(feats, idx, p, pe1W, pe1b, pe2W, pe2b, out, N, D, H, Wd, B, stream);
      outOff += (size_t)N * 64;
    }
  }
}
